// Linear_int_90950227460261
// MI455X (gfx1250) — 8-bit matrix operand order repaired and checked by reading, compiled, not run on hardware
//
#include <hip/hip_runtime.h>
#include <cstdint>
#include <cstddef>


typedef __attribute__((ext_vector_type(8))) int v8i;

#define M_TOTAL 4096
#define K_DIM   4096
#define N_DIM   4096

#define WMMA_IU8(Afrag, Bfrag, Cfrag) \
    __builtin_amdgcn_wmma_i32_16x16x64_iu8(true, (Afrag), true, (Bfrag), (Cfrag), false, false)

__global__ __launch_bounds__(256)
void pack_i8_kernel(const int* __restrict__ src, int8_t* __restrict__ dst, int n4)
{
    int i = blockIdx.x * blockDim.x + threadIdx.x;
    if (i < n4) {
        int4 v = ((const int4*)src)[i];
        char4 p;
        p.x = (char)v.x;
        p.y = (char)v.y;
        p.z = (char)v.z;
        p.w = (char)v.w;
        ((char4*)dst)[i] = p;
    }
}

__device__ __forceinline__ v8i load_a_frag(const int8_t* p)
{
    int4 a0 = *(const int4*)(p);
    int4 a1 = *(const int4*)(p + 32);
    v8i v;
    v[0] = a0.x; v[1] = a0.y; v[2] = a0.z; v[3] = a0.w;
    v[4] = a1.x; v[5] = a1.y; v[6] = a1.z; v[7] = a1.w;
    return v;
}

__device__ __forceinline__ v8i load_b_frag(const int8_t* p)
{
    int4 b0 = *(const int4*)(p);
    int4 b1 = *(const int4*)(p + 32);
    v8i v;
    v[0] = b0.x; v[1] = b0.y; v[2] = b0.z; v[3] = b0.w;
    v[4] = b1.x; v[5] = b1.y; v[6] = b1.z; v[7] = b1.w;
    return v;
}

__device__ __forceinline__ void store_tile(v8i acc, int m0, int n0,
                                           int half, int lane_lo,
                                           float* __restrict__ q_out,
                                           float* __restrict__ r_out)
{
    #pragma unroll
    for (int i = 0; i < 8; ++i) {
        const int m = m0 + i + half * 8;
        const int n = n0 + lane_lo;
        const int a = acc[i];
        const size_t idx = (size_t)m * N_DIM + n;
        q_out[idx] = (float)(a >> 12);
        r_out[idx] = (float)(a & 4095);
    }
}

__global__ __launch_bounds__(256)
void linear_int_wmma_kernel(const int8_t* __restrict__ A,
                            const int8_t* __restrict__ W,
                            float* __restrict__ q_out,
                            float* __restrict__ r_out)
{
    const int wave    = threadIdx.x >> 5;
    const int lane    = threadIdx.x & 31;
    const int lane_lo = lane & 15;
    const int half    = lane >> 4;

    const int wave_m = wave >> 2;
    const int wave_n = wave & 3;

    const int m_base = blockIdx.y * 64  + wave_m * 32;
    const int n_base = blockIdx.x * 128 + wave_n * 32;

    const int8_t* aptr0 = A + (size_t)(m_base + lane_lo) * K_DIM + half * 16;
    const int8_t* aptr1 = aptr0 + (size_t)16 * K_DIM;
    const int8_t* bptr0 = W + (size_t)(n_base + lane_lo) * K_DIM + half * 16;
    const int8_t* bptr1 = bptr0 + (size_t)16 * K_DIM;

    v8i acc00 = {0, 0, 0, 0, 0, 0, 0, 0};
    v8i acc01 = {0, 0, 0, 0, 0, 0, 0, 0};
    v8i acc10 = {0, 0, 0, 0, 0, 0, 0, 0};
    v8i acc11 = {0, 0, 0, 0, 0, 0, 0, 0};

    v8i a0 = load_a_frag(aptr0);
    v8i a1 = load_a_frag(aptr1);
    v8i b0 = load_b_frag(bptr0);
    v8i b1 = load_b_frag(bptr1);

    #pragma unroll 1
    for (int k = 0; k < K_DIM; k += 128) {
        const v8i xa0 = load_a_frag(aptr0 + (k + 64));
        const v8i xa1 = load_a_frag(aptr1 + (k + 64));
        const v8i xb0 = load_b_frag(bptr0 + (k + 64));
        const v8i xb1 = load_b_frag(bptr1 + (k + 64));

        const int kp = (k + 256) & (K_DIM - 1);
        __builtin_prefetch(aptr0 + kp, 0, 3);
        __builtin_prefetch(aptr1 + kp, 0, 3);
        __builtin_prefetch(bptr0 + kp, 0, 3);
        __builtin_prefetch(bptr1 + kp, 0, 3);

        acc00 = WMMA_IU8(a0, b0, acc00);
        acc01 = WMMA_IU8(a0, b1, acc01);
        acc10 = WMMA_IU8(a1, b0, acc10);
        acc11 = WMMA_IU8(a1, b1, acc11);

        const int kn = (k + 128) & (K_DIM - 1);
        a0 = load_a_frag(aptr0 + kn);
        a1 = load_a_frag(aptr1 + kn);
        b0 = load_b_frag(bptr0 + kn);
        b1 = load_b_frag(bptr1 + kn);

        acc00 = WMMA_IU8(xa0, xb0, acc00);
        acc01 = WMMA_IU8(xa0, xb1, acc01);
        acc10 = WMMA_IU8(xa1, xb0, acc10);
        acc11 = WMMA_IU8(xa1, xb1, acc11);
    }

    store_tile(acc00, m_base,      n_base,      half, lane_lo, q_out, r_out);
    store_tile(acc01, m_base,      n_base + 16, half, lane_lo, q_out, r_out);
    store_tile(acc10, m_base + 16, n_base,      half, lane_lo, q_out, r_out);
    store_tile(acc11, m_base + 16, n_base + 16, half, lane_lo, q_out, r_out);
}

extern "C" void kernel_launch(void* const* d_in, const int* in_sizes, int n_in,
                              void* d_out, int out_size, void* d_ws, size_t ws_size,
                              hipStream_t stream)
{
    (void)in_sizes; (void)n_in; (void)out_size; (void)ws_size;

    const int* x = (const int*)d_in[0];
    const int* w = (const int*)d_in[1];

    int8_t* x8 = (int8_t*)d_ws;
    int8_t* w8 = x8 + (size_t)M_TOTAL * K_DIM;

    float* q_out = (float*)d_out;
    float* r_out = q_out + (size_t)M_TOTAL * N_DIM;

    {
        const int n4x = (M_TOTAL * K_DIM) / 4;
        pack_i8_kernel<<<(n4x + 255) / 256, 256, 0, stream>>>(x, x8, n4x);
        const int n4w = (N_DIM * K_DIM) / 4;
        pack_i8_kernel<<<(n4w + 255) / 256, 256, 0, stream>>>(w, w8, n4w);
    }

    dim3 grid(N_DIM / 128, M_TOTAL / 64);
    linear_int_wmma_kernel<<<grid, 256, 0, stream>>>(x8, w8, q_out, r_out);
}
